// SwinDecoder_68934225100964
// MI455X (gfx1250) — compile-verified, matrix operand order checked by reading, not run on hardware
//
#include <hip/hip_runtime.h>

// ---------------------------------------------------------------------------
// Swin decoder stage for gfx1250 (MI455X), wave32.
// All GEMMs + attention on v_wmma_f32_16x16x32_bf16, bf16 activations
// end-to-end (fp32 accumulate, fp32 residual spine). Weights are converted
// once per launch to transposed [N][K] bf16 so *both* GEMM staging paths are
// pure 128-bit global->reg->LDS copies (register double-buffered) + prefetch.
// ---------------------------------------------------------------------------

typedef __bf16 bf16_t;
typedef bf16_t v4bf  __attribute__((ext_vector_type(4)));
typedef bf16_t v8bf  __attribute__((ext_vector_type(8)));
typedef bf16_t v16bf __attribute__((ext_vector_type(16)));
typedef float  v8f   __attribute__((ext_vector_type(8)));

static constexpr int Bn = 16, Hh = 56, Ww = 56, Dd = 384;
static constexpr int NH = 12, WS = 7, SH = 3, HD = 32;
static constexpr int Nn = WS * WS;                 // 49
static constexpr int nW = (Hh / WS) * (Ww / WS);   // 64
static constexpr int Ll = Hh * Ww;                 // 3136
static constexpr int ROWS = Bn * Ll;               // 50176 == B*nW*N

__device__ __forceinline__ bf16_t f2bf(float f) {
    unsigned u = __builtin_bit_cast(unsigned, f);
    unsigned r = u + 0x7FFFu + ((u >> 16) & 1u);   // round-to-nearest-even
    return __builtin_bit_cast(bf16_t, (unsigned short)(r >> 16));
}
__device__ __forceinline__ bf16_t bf_zero() {
    return __builtin_bit_cast(bf16_t, (unsigned short)0);
}

// Load one 16-bit A/B WMMA fragment from LDS. The 32 K-values for this
// row/column are contiguous at `base`; lane half (l>>4) owns K chunks
// {c0..c0+7} and {16+c0..16+c0+7} per the CDNA5 16-bit operand layout.
__device__ __forceinline__ v16bf ld_frag(const bf16_t* base, int c0) {
    union { v16bf v; v8bf h[2]; } u;
    u.h[0] = *(const v8bf*)(base + c0);
    u.h[1] = *(const v8bf*)(base + 16 + c0);
    return u.v;
}

// ---------------------------------------------------------------------------
// fp32 -> bf16 bulk convert (vectorized x4)
// ---------------------------------------------------------------------------
__global__ __launch_bounds__(256) void cvt_bf16_kernel(
    const float* __restrict__ in, bf16_t* __restrict__ out, int n4)
{
    int i = blockIdx.x * 256 + threadIdx.x;
    if (i < n4) {
        float4 v = ((const float4*)in)[i];
        v4bf o = { f2bf(v.x), f2bf(v.y), f2bf(v.z), f2bf(v.w) };
        ((v4bf*)out)[i] = o;
    }
}

// ---------------------------------------------------------------------------
// fp32 [K][N] -> bf16 [N][K] transposed convert (LDS-tiled, coalesced both
// sides). Runs once per launch over the weights. Requires K%32==0, N%32==0.
// ---------------------------------------------------------------------------
__global__ __launch_bounds__(256) void cvt_tr_kernel(
    const float* __restrict__ in, bf16_t* __restrict__ out, int K, int N)
{
    __shared__ float tile[32][33];
    const int bk = blockIdx.y * 32, bn = blockIdx.x * 32;
    const int tx = threadIdx.x & 31, ty = threadIdx.x >> 5;  // 32 x 8
    #pragma unroll
    for (int j = 0; j < 32; j += 8)
        tile[ty + j][tx] = in[(size_t)(bk + ty + j) * N + bn + tx];
    __syncthreads();
    #pragma unroll
    for (int j = 0; j < 32; j += 8)
        out[(size_t)(bn + ty + j) * K + bk + tx] = f2bf(tile[tx][ty + j]);
}

// ---------------------------------------------------------------------------
// LayerNorm (D=384), bf16 output, optional fused window partition (+shift).
// mode: 0 = identity rows, 1 = partition, 2 = partition with roll(-SH).
// ---------------------------------------------------------------------------
__global__ __launch_bounds__(128) void ln_part_kernel(
    const float* __restrict__ x, const float* __restrict__ gamma,
    const float* __restrict__ beta, bf16_t* __restrict__ out, int mode)
{
    __shared__ float red[128];
    const int r = blockIdx.x;
    const int tid = threadIdx.x;
    const float* xr = x + (size_t)r * Dd;
    float v0 = xr[tid], v1 = xr[tid + 128], v2 = xr[tid + 256];

    red[tid] = v0 + v1 + v2; __syncthreads();
    for (int s = 64; s > 0; s >>= 1) { if (tid < s) red[tid] += red[tid + s]; __syncthreads(); }
    float mean = red[0] * (1.0f / Dd);
    __syncthreads();
    float d0 = v0 - mean, d1 = v1 - mean, d2 = v2 - mean;
    red[tid] = d0 * d0 + d1 * d1 + d2 * d2; __syncthreads();
    for (int s = 64; s > 0; s >>= 1) { if (tid < s) red[tid] += red[tid + s]; __syncthreads(); }
    float rstd = rsqrtf(red[0] * (1.0f / Dd) + 1e-5f);

    int orow = r;
    if (mode) {
        int b = r / Ll, hw = r % Ll, h = hw / Ww, w = hw % Ww;
        int h2 = h, w2 = w;
        if (mode == 2) { h2 = (h + Hh - SH) % Hh; w2 = (w + Ww - SH) % Ww; }
        int win = (h2 / WS) * (Ww / WS) + (w2 / WS);
        int n   = (h2 % WS) * WS + (w2 % WS);
        orow = (b * nW + win) * Nn + n;
    }
    bf16_t* op = out + (size_t)orow * Dd;
    op[tid]       = f2bf(d0 * rstd * gamma[tid]       + beta[tid]);
    op[tid + 128] = f2bf(d1 * rstd * gamma[tid + 128] + beta[tid + 128]);
    op[tid + 256] = f2bf(d2 * rstd * gamma[tid + 256] + beta[tid + 256]);
}

// ---------------------------------------------------------------------------
// x += window_reverse(op) with optional roll(+SH) (inverse of shift).
// ---------------------------------------------------------------------------
__global__ __launch_bounds__(128) void add_rev_kernel(
    float* __restrict__ x, const float* __restrict__ op, int shifted)
{
    const int r = blockIdx.x, tid = threadIdx.x;
    int b = r / Ll, hw = r % Ll, h = hw / Ww, w = hw % Ww;
    int hh = h, ww = w;
    if (shifted) { hh = (h + Hh - SH) % Hh; ww = (w + Ww - SH) % Ww; }
    int orow = (b * nW + (hh / WS) * (Ww / WS) + (ww / WS)) * Nn + (hh % WS) * WS + (ww % WS);
    const float* s = op + (size_t)orow * Dd;
    float* d = x + (size_t)r * Dd;
    d[tid]       += s[tid];
    d[tid + 128] += s[tid + 128];
    d[tid + 256] += s[tid + 256];
}

// ---------------------------------------------------------------------------
// Tiled bf16 WMMA GEMM: C[M,N] = epilogue(A[M,K] @ Wt[N][K]^T).
// Block 256 thr (8 waves), tile 128x128, wave computes 32x64 (2x4 WMMA tiles).
// A and Wt are both K-contiguous => staging is pure b128 copies, register
// double-buffered, with L2 prefetch two K-tiles ahead.
// Assumes M%128==0, N%128==0, K%32==0 (true for every call here).
// ---------------------------------------------------------------------------
#define GF_BIAS   1
#define GF_GELU   2
#define GF_RESID  4
#define GF_OUTBF  8

__global__ __launch_bounds__(256) void gemm_kernel(
    const bf16_t* __restrict__ A, const bf16_t* __restrict__ Wt,
    const float* __restrict__ bias, const float* __restrict__ resid,
    void* __restrict__ Cout, int M, int N, int K, int flags)
{
    __shared__ bf16_t lA[128 * 32];    // A tile   [row][k]
    __shared__ bf16_t lBt[128 * 32];   // Wt tile  [n][k]
    const int tid  = threadIdx.x;
    const int lane = tid & 31;
    const int wave = tid >> 5;
    const int bm = blockIdx.y * 128;
    const int bn = blockIdx.x * 128;
    const int waveM = (wave & 3) * 32;
    const int waveN = (wave >> 2) * 64;
    const int c0 = (lane >> 4) * 8;
    const int srr = tid >> 2, scq = (tid & 3) * 8;   // staging coords (f = tid, tid+256 => +64 rows)

    v8f acc[2][4] = {};
    v8bf ra[2], rw[2];

    auto gload = [&](int k0) {
        #pragma unroll
        for (int i = 0; i < 2; ++i)
            ra[i] = *(const v8bf*)(A  + (size_t)(bm + srr + i * 64) * K + k0 + scq);
        #pragma unroll
        for (int i = 0; i < 2; ++i)
            rw[i] = *(const v8bf*)(Wt + (size_t)(bn + srr + i * 64) * K + k0 + scq);
    };

    gload(0);
    for (int k0 = 0; k0 < K; k0 += 32) {
        __syncthreads();                            // LDS free from last iter
        #pragma unroll
        for (int i = 0; i < 2; ++i)
            *(v8bf*)&lA[(srr + i * 64) * 32 + scq] = ra[i];
        #pragma unroll
        for (int i = 0; i < 2; ++i)
            *(v8bf*)&lBt[(srr + i * 64) * 32 + scq] = rw[i];
        __syncthreads();

        if (k0 + 32 < K) {
            gload(k0 + 32);                          // overlap with WMMA burst
            if (k0 + 96 < K) {
                __builtin_prefetch(A  + (size_t)(bm + (tid >> 1)) * K + k0 + 96, 0, 1);
                __builtin_prefetch(Wt + (size_t)(bn + (tid >> 1)) * K + k0 + 96, 0, 1);
            }
        }

        v16bf af[2], bfr[4];
        #pragma unroll
        for (int mt = 0; mt < 2; ++mt)
            af[mt] = ld_frag(&lA[(waveM + mt * 16 + (lane & 15)) * 32], c0);
        #pragma unroll
        for (int nt = 0; nt < 4; ++nt)
            bfr[nt] = ld_frag(&lBt[(waveN + nt * 16 + (lane & 15)) * 32], c0);
        #pragma unroll
        for (int mt = 0; mt < 2; ++mt)
            #pragma unroll
            for (int nt = 0; nt < 4; ++nt)
                acc[mt][nt] = __builtin_amdgcn_wmma_f32_16x16x32_bf16(
                    false, af[mt], false, bfr[nt], (short)0, acc[mt][nt], false, false);
    }

    // epilogue: C-layout => lane holds col (lane&15), rows i + 8*(lane>>4)
    #pragma unroll
    for (int mt = 0; mt < 2; ++mt)
        #pragma unroll
        for (int nt = 0; nt < 4; ++nt) {
            int gc = bn + waveN + nt * 16 + (lane & 15);
            float bv = (flags & GF_BIAS) ? bias[gc] : 0.0f;
            #pragma unroll
            for (int i = 0; i < 8; ++i) {
                int gr = bm + waveM + mt * 16 + (lane >> 4) * 8 + i;
                float v = acc[mt][nt][i] + bv;
                if (flags & GF_GELU)  v = 0.5f * v * (1.0f + erff(v * 0.70710678118654752f));
                if (flags & GF_RESID) v += resid[(size_t)gr * N + gc];
                if (flags & GF_OUTBF) ((bf16_t*)Cout)[(size_t)gr * N + gc] = f2bf(v);
                else                  ((float*)Cout)[(size_t)gr * N + gc] = v;
            }
        }
}

// ---------------------------------------------------------------------------
// Fused window attention: one wave per (batch, window, head).
// S = Q K^T (padded 64x64), logits = S*HD^-0.5 + rel-pos bias + shift mask,
// softmax, O = P V. Both GEMMs via WMMA bf16; qkv/O buffers are bf16.
// ---------------------------------------------------------------------------
__global__ __launch_bounds__(32) void attn_kernel(
    const bf16_t* __restrict__ qkv, const float* __restrict__ rpb,
    bf16_t* __restrict__ O, int shifted)
{
    __shared__ bf16_t lQ[64 * 32];   // [row][d]
    __shared__ bf16_t lK[64 * 32];   // [key][d]
    __shared__ bf16_t lVt[32 * 64];  // [d][key]
    __shared__ bf16_t lP[64 * 64];   // softmax probs [row][key]

    const int lane = threadIdx.x;
    int idx = blockIdx.x;
    const int head = idx % NH; idx /= NH;
    const int win  = idx % nW; const int b = idx / nW;
    const size_t rowbase = (size_t)(b * nW + win) * Nn;
    const float qscale = 0.17677669529663687f;  // HD^-0.5, folded into logits

    const bf16_t z0 = bf_zero();

    // ---- stage Q,K,V (pad rows 49..63 with zeros); pure bf16 copies ----
    #pragma unroll
    for (int t = 0; t < 2; ++t) {
        int rr = lane + t * 32;
        if (rr < Nn) {
            const bf16_t* p = qkv + (rowbase + rr) * (3 * Dd) + head * HD;
            #pragma unroll
            for (int j = 0; j < 4; ++j) {
                v8bf q = *(const v8bf*)(p + j * 8);
                v8bf k = *(const v8bf*)(p + Dd + j * 8);
                v8bf v = *(const v8bf*)(p + 2 * Dd + j * 8);
                *(v8bf*)&lQ[rr * 32 + j * 8] = q;
                *(v8bf*)&lK[rr * 32 + j * 8] = k;
                #pragma unroll
                for (int e = 0; e < 8; ++e) lVt[(j * 8 + e) * 64 + rr] = v[e];
            }
        } else {
            v8bf z = { z0, z0, z0, z0, z0, z0, z0, z0 };
            #pragma unroll
            for (int j = 0; j < 4; ++j) {
                *(v8bf*)&lQ[rr * 32 + j * 8] = z;
                *(v8bf*)&lK[rr * 32 + j * 8] = z;
            }
            #pragma unroll
            for (int d = 0; d < 32; ++d) lVt[d * 64 + rr] = z0;
        }
    }
    __syncthreads();

    const int c0 = (lane >> 4) * 8;

    // ---- S = Q K^T : 4x4 tiles, K-dim = 32 (one WMMA each) ----
    v8f S[4][4] = {};
    v16bf kf[4];
    #pragma unroll
    for (int nt = 0; nt < 4; ++nt)
        kf[nt] = ld_frag(&lK[(nt * 16 + (lane & 15)) * 32], c0);
    #pragma unroll
    for (int mt = 0; mt < 4; ++mt) {
        v16bf qf = ld_frag(&lQ[(mt * 16 + (lane & 15)) * 32], c0);
        #pragma unroll
        for (int nt = 0; nt < 4; ++nt)
            S[mt][nt] = __builtin_amdgcn_wmma_f32_16x16x32_bf16(
                false, qf, false, kf[nt], (short)0, S[mt][nt], false, false);
    }

    // ---- scale + bias + mask + row softmax (rows live in 16-lane halves) ----
    const int wh = win / (Ww / WS), wc = win % (Ww / WS);
    #pragma unroll
    for (int mt = 0; mt < 4; ++mt) {
        #pragma unroll
        for (int i = 0; i < 8; ++i) {
            int a = mt * 16 + (lane >> 4) * 8 + i;      // query index (padded)
            int ia = a / WS, ja = a % WS;
            int rga = 0;
            if (shifted && a < Nn) {
                int hA = wh * WS + ia, wA = wc * WS + ja;
                rga = (hA < Hh - WS ? 0 : (hA < Hh - SH ? 1 : 2)) * 3 +
                      (wA < Ww - WS ? 0 : (wA < Ww - SH ? 1 : 2));
            }
            float sv[4], m = -1e30f;
            #pragma unroll
            for (int nt = 0; nt < 4; ++nt) {
                int cidx = nt * 16 + (lane & 15);
                float s = S[mt][nt][i] * qscale;
                if (a < Nn && cidx < Nn) {
                    int ic = cidx / WS, jc = cidx % WS;
                    s += rpb[((ia - ic + WS - 1) * (2 * WS - 1) + (ja - jc + WS - 1)) * NH + head];
                    if (shifted) {
                        int hC = wh * WS + ic, wC = wc * WS + jc;
                        int rgc = (hC < Hh - WS ? 0 : (hC < Hh - SH ? 1 : 2)) * 3 +
                                  (wC < Ww - WS ? 0 : (wC < Ww - SH ? 1 : 2));
                        if (rgc != rga) s += -100.0f;
                    }
                } else {
                    s = -1e30f;
                }
                sv[nt] = s; m = fmaxf(m, s);
            }
            #pragma unroll
            for (int off = 8; off > 0; off >>= 1) m = fmaxf(m, __shfl_xor(m, off, 32));
            float sum = 0.0f;
            #pragma unroll
            for (int nt = 0; nt < 4; ++nt) { sv[nt] = __expf(sv[nt] - m); sum += sv[nt]; }
            #pragma unroll
            for (int off = 8; off > 0; off >>= 1) sum += __shfl_xor(sum, off, 32);
            float inv = 1.0f / (sum + 1e-20f);
            #pragma unroll
            for (int nt = 0; nt < 4; ++nt)
                lP[a * 64 + nt * 16 + (lane & 15)] = f2bf(sv[nt] * inv);
        }
    }
    __syncthreads();

    // ---- O = P V : 4x2 tiles over K=64 (two 32-steps) ----
    v8f Oacc[4][2] = {};
    #pragma unroll
    for (int kt = 0; kt < 2; ++kt) {
        v16bf vf[2];
        #pragma unroll
        for (int nt = 0; nt < 2; ++nt)
            vf[nt] = ld_frag(&lVt[(nt * 16 + (lane & 15)) * 64 + kt * 32], c0);
        #pragma unroll
        for (int mt = 0; mt < 4; ++mt) {
            v16bf pf = ld_frag(&lP[(mt * 16 + (lane & 15)) * 64 + kt * 32], c0);
            #pragma unroll
            for (int nt = 0; nt < 2; ++nt)
                Oacc[mt][nt] = __builtin_amdgcn_wmma_f32_16x16x32_bf16(
                    false, pf, false, vf[nt], (short)0, Oacc[mt][nt], false, false);
        }
    }
    #pragma unroll
    for (int mt = 0; mt < 4; ++mt)
        #pragma unroll
        for (int nt = 0; nt < 2; ++nt)
            #pragma unroll
            for (int i = 0; i < 8; ++i) {
                int a = mt * 16 + (lane >> 4) * 8 + i;
                if (a < Nn) {
                    int d = nt * 16 + (lane & 15);
                    O[(rowbase + a) * Dd + head * HD + d] = f2bf(Oacc[mt][nt][i]);
                }
            }
}

// ---------------------------------------------------------------------------
// PatchExpand pixel-shuffle + final LayerNorm(192).
// ---------------------------------------------------------------------------
__global__ __launch_bounds__(64) void expand_ln_kernel(
    const float* __restrict__ XE, const float* __restrict__ g,
    const float* __restrict__ be, float* __restrict__ out)
{
    __shared__ float red[64];
    const int rout = blockIdx.x, tid = threadIdx.x;
    int b = rout / (4 * Ll), rr = rout % (4 * Ll);
    int h = rr / 224, r2 = rr % 224;
    int p = r2 / 112, r3 = r2 % 112;
    int w = r3 >> 1, q = r3 & 1;
    const float* src = XE + ((size_t)(b * Ll + h * Ww + w)) * (2 * Dd) + (p * 2 + q) * 192;

    float v0 = src[tid], v1 = src[tid + 64], v2 = src[tid + 128];
    red[tid] = v0 + v1 + v2; __syncthreads();
    for (int s = 32; s > 0; s >>= 1) { if (tid < s) red[tid] += red[tid + s]; __syncthreads(); }
    float mean = red[0] * (1.0f / 192.0f);
    __syncthreads();
    float d0 = v0 - mean, d1 = v1 - mean, d2 = v2 - mean;
    red[tid] = d0 * d0 + d1 * d1 + d2 * d2; __syncthreads();
    for (int s = 32; s > 0; s >>= 1) { if (tid < s) red[tid] += red[tid + s]; __syncthreads(); }
    float rstd = rsqrtf(red[0] * (1.0f / 192.0f) + 1e-5f);

    float* op = out + (size_t)rout * 192;
    op[tid]       = d0 * rstd * g[tid]       + be[tid];
    op[tid + 64]  = d1 * rstd * g[tid + 64]  + be[tid + 64];
    op[tid + 128] = d2 * rstd * g[tid + 128] + be[tid + 128];
}

// ---------------------------------------------------------------------------
extern "C" void kernel_launch(void* const* d_in, const int* in_sizes, int n_in,
                              void* d_out, int out_size, void* d_ws, size_t ws_size,
                              hipStream_t stream)
{
    const float* x_in     = (const float*)d_in[0];
    const float* norm1_g  = (const float*)d_in[1];
    const float* norm1_b  = (const float*)d_in[2];
    const float* qkv_w    = (const float*)d_in[3];
    const float* qkv_b    = (const float*)d_in[4];
    const float* rpb      = (const float*)d_in[5];
    const float* proj_w   = (const float*)d_in[6];
    const float* proj_b   = (const float*)d_in[7];
    const float* norm2_g  = (const float*)d_in[8];
    const float* norm2_b  = (const float*)d_in[9];
    const float* fc1_w    = (const float*)d_in[10];
    const float* fc1_b    = (const float*)d_in[11];
    const float* fc2_w    = (const float*)d_in[12];
    const float* fc2_b    = (const float*)d_in[13];
    const float* expand_w = (const float*)d_in[14];
    const float* exnorm_g = (const float*)d_in[15];
    const float* exnorm_b = (const float*)d_in[16];

    // ---- workspace layout ----
    char* ws = (char*)d_ws;
    const size_t szX   = (size_t)ROWS * Dd * sizeof(float);        // 77 MB fp32 spine
    const size_t szLbf = (size_t)ROWS * Dd * sizeof(bf16_t);       // 38.5 MB
    const size_t szQbf = (size_t)ROWS * 3 * Dd * sizeof(bf16_t);   // 115.6 MB (also fp32 proj out)
    const size_t szH   = (size_t)ROWS * 4 * Dd * sizeof(bf16_t);   // 154 MB (== ROWS*2D fp32)
    float*  X     = (float*)(ws);
    bf16_t* bufL  = (bf16_t*)(ws + szX);
    bf16_t* bufQ  = (bf16_t*)(ws + szX + szLbf);
    bf16_t* bufH  = (bf16_t*)(ws + szX + szLbf + szQbf);
    bf16_t* wts   = (bf16_t*)(ws + szX + szLbf + szQbf + szH);

    // transposed bf16 weight slots ([N][K] each slice)
    bf16_t* qkvW = wts;                           // 2 x (3D x D)
    bf16_t* prjW = qkvW + 2 * Dd * 3 * Dd;        // 2 x (D x D)
    bf16_t* f1W  = prjW + 2 * Dd * Dd;            // 2 x (4D x D)
    bf16_t* f2W  = f1W  + 2 * Dd * 4 * Dd;        // 2 x (D x 4D)
    bf16_t* expW = f2W  + 2 * 4 * Dd * Dd;        // 2D x D

    auto cvt_tr = [&](const float* src, bf16_t* dst, int K, int N) {
        cvt_tr_kernel<<<dim3(N / 32, K / 32), 256, 0, stream>>>(src, dst, K, N);
    };
    for (int i = 0; i < 2; ++i) {
        cvt_tr(qkv_w  + (size_t)i * Dd * 3 * Dd, qkvW + (size_t)i * 3 * Dd * Dd, Dd, 3 * Dd);
        cvt_tr(proj_w + (size_t)i * Dd * Dd,     prjW + (size_t)i * Dd * Dd,     Dd, Dd);
        cvt_tr(fc1_w  + (size_t)i * Dd * 4 * Dd, f1W  + (size_t)i * 4 * Dd * Dd, Dd, 4 * Dd);
        cvt_tr(fc2_w  + (size_t)i * 4 * Dd * Dd, f2W  + (size_t)i * Dd * 4 * Dd, 4 * Dd, Dd);
    }
    cvt_tr(expand_w, expW, Dd, 2 * Dd);

    hipMemcpyAsync(X, x_in, szX, hipMemcpyDeviceToDevice, stream);

    const dim3 blkLN(128), blkG(256), blkA(32), blkE(64);

    for (int i = 0; i < 2; ++i) {
        const int shifted = i;  // block 0: W-MSA, block 1: SW-MSA
        // LN1 fused with window partition (+shift), bf16 out
        ln_part_kernel<<<ROWS, blkLN, 0, stream>>>(
            X, norm1_g + i * Dd, norm1_b + i * Dd, bufL, shifted ? 2 : 1);
        // qkv GEMM -> bf16
        gemm_kernel<<<dim3(3 * Dd / 128, ROWS / 128), blkG, 0, stream>>>(
            bufL, qkvW + (size_t)i * 3 * Dd * Dd, qkv_b + i * 3 * Dd, nullptr,
            bufQ, ROWS, 3 * Dd, Dd, GF_BIAS | GF_OUTBF);
        // fused windowed attention (bf16 in/out)
        attn_kernel<<<Bn * nW * NH, blkA, 0, stream>>>(
            bufQ, rpb + (size_t)i * (2 * WS - 1) * (2 * WS - 1) * NH, bufL, shifted);
        // proj GEMM -> fp32 (into bufQ region)
        gemm_kernel<<<dim3(Dd / 128, ROWS / 128), blkG, 0, stream>>>(
            bufL, prjW + (size_t)i * Dd * Dd, proj_b + i * Dd, nullptr,
            (float*)bufQ, ROWS, Dd, Dd, GF_BIAS);
        // residual + window reverse (+unshift)
        add_rev_kernel<<<ROWS, blkLN, 0, stream>>>(X, (const float*)bufQ, shifted);
        // LN2 -> bf16
        ln_part_kernel<<<ROWS, blkLN, 0, stream>>>(
            X, norm2_g + i * Dd, norm2_b + i * Dd, bufL, 0);
        // fc1 + GELU -> bf16
        gemm_kernel<<<dim3(4 * Dd / 128, ROWS / 128), blkG, 0, stream>>>(
            bufL, f1W + (size_t)i * 4 * Dd * Dd, fc1_b + i * 4 * Dd, nullptr,
            bufH, ROWS, 4 * Dd, Dd, GF_BIAS | GF_GELU | GF_OUTBF);
        // fc2 + residual -> X (fp32, in-place residual: read-once/write-once)
        gemm_kernel<<<dim3(Dd / 128, ROWS / 128), blkG, 0, stream>>>(
            bufH, f2W + (size_t)i * Dd * 4 * Dd, fc2_b + i * Dd, X,
            X, ROWS, Dd, 4 * Dd, GF_BIAS | GF_RESID);
    }

    // PatchExpand GEMM: bf16(X) @ expand_w -> fp32 (bufH region)
    {
        int n4 = (int)((size_t)ROWS * Dd / 4);
        cvt_bf16_kernel<<<(n4 + 255) / 256, 256, 0, stream>>>(X, bufL, n4);
    }
    gemm_kernel<<<dim3(2 * Dd / 128, ROWS / 128), blkG, 0, stream>>>(
        bufL, expW, nullptr, nullptr, (float*)bufH, ROWS, 2 * Dd, Dd, 0);
    // pixel shuffle + final LN(192)
    expand_ln_kernel<<<Bn * 4 * Ll, blkE, 0, stream>>>(
        (const float*)bufH, exnorm_g, exnorm_b, (float*)d_out);
    (void)in_sizes; (void)n_in; (void)out_size; (void)ws_size;
}
